// EnhancedFeatureExtractor_75969381531891
// MI455X (gfx1250) — matrix operand order repaired and checked by reading, not compiled, not run on hardware
//
#include <hip/hip_runtime.h>
#include <math.h>

#define D_MODEL 768
#define NHEADS  12
#define HDIM    64
#define BATCH   32
#define SEQ     512
#define CTX     128

typedef __attribute__((ext_vector_type(16))) __bf16 v16bf;
typedef __attribute__((ext_vector_type(8)))  __bf16 v8bf;
typedef __attribute__((ext_vector_type(8)))  float  v8f;

__device__ __forceinline__ v8f wmma_bf16(v16bf a, v16bf b, v8f c) {
#if defined(__gfx1250__)
  // D = A(16x32 bf16) * B(32x16 bf16) + C(16x16 f32)
  return __builtin_amdgcn_wmma_f32_16x16x32_bf16(false, a, false, b, (short)0, c,
                                                 false, false);
#else
  (void)a; (void)b;
  return c;
#endif
}

// Combine the two 8-element halves of an A fragment (K = base..base+7 and
// K = base+8..base+15; contiguous K order, matching the B fragments).
__device__ __forceinline__ v16bf frag_from_halves(v8bf lo, v8bf hi) {
  return __builtin_shufflevector(lo, hi, 0, 1, 2, 3, 4, 5, 6, 7,
                                 8, 9, 10, 11, 12, 13, 14, 15);
}

// ---------------------------------------------------------------------------
// f32 -> bf16 conversion (one pass per tensor; keeps cvts out of GEMM loop)
// ---------------------------------------------------------------------------
__global__ void efx_cvt_bf16(const float* __restrict__ X, __bf16* __restrict__ Y,
                             int n)
{
  const int i = blockIdx.x * 256 + threadIdx.x;
  if (i < n) Y[i] = (__bf16)X[i];
}

// ---------------------------------------------------------------------------
// GEMM: Y[M,N] = X[M,K] @ W[N,K]^T + bias[N]  (optional ReLU)
// X, W pre-converted bf16 -> inner loop is pure b128 loads + WMMA.
// Wave tile 32(M) x 64(N): every B fragment feeds two WMMAs -> 1.5 loads/WMMA.
// Block = 128 threads = 4 waves stacked in M -> block tile 128 x 64.
// ---------------------------------------------------------------------------
__global__ __launch_bounds__(128)
void efx_gemm_bias(const __bf16* __restrict__ X, const __bf16* __restrict__ W,
                   const float* __restrict__ bias,
                   float* __restrict__ Yf, __bf16* __restrict__ Yh,
                   int M, int N, int K, int relu)
{
  const int lane  = threadIdx.x & 31;
  const int wave  = threadIdx.x >> 5;
  const int laneN = lane & 15;
  const int g     = lane >> 4;          // lane half: selects K sub-ranges
  const int m0    = blockIdx.y * 128 + wave * 32;
  const int n0    = blockIdx.x * 64;

  int rowA0 = m0 + laneN;               // A-fragment rows (M indices)
  int rowA1 = m0 + 16 + laneN;
  if (rowA0 >= M) rowA0 = M - 1;        // clamp loads; stores are guarded
  if (rowA1 >= M) rowA1 = M - 1;

  const v8f vzero = {0.f,0.f,0.f,0.f,0.f,0.f,0.f,0.f};
  v8f acc0[4], acc1[4];
#pragma unroll
  for (int t = 0; t < 4; ++t) { acc0[t] = vzero; acc1[t] = vzero; }

  for (int k0 = 0; k0 < K; k0 += 32) {
    // Two A fragments (rows m0..m0+15 and m0+16..m0+31)
    const __bf16* xp0 = X + (size_t)rowA0 * K + k0 + g * 16;
    const __bf16* xp1 = X + (size_t)rowA1 * K + k0 + g * 16;
    v16bf a0 = frag_from_halves(*(const v8bf*)xp0, *(const v8bf*)(xp0 + 8));
    v16bf a1 = frag_from_halves(*(const v8bf*)xp1, *(const v8bf*)(xp1 + 8));
#pragma unroll
    for (int nt = 0; nt < 4; ++nt) {
      // B fragment: B[k][n] = W[n][k]; 16 contiguous bf16 (32B aligned)
      const int col = n0 + nt * 16 + laneN;
      v16bf bb = *(const v16bf*)(W + (size_t)col * K + k0 + g * 16);
      acc0[nt] = wmma_bf16(a0, bb, acc0[nt]);
      acc1[nt] = wmma_bf16(a1, bb, acc1[nt]);
    }
  }

#pragma unroll
  for (int nt = 0; nt < 4; ++nt) {
    const int col  = n0 + nt * 16 + laneN;
    const float bv = bias ? bias[col] : 0.0f;
#pragma unroll
    for (int r = 0; r < 8; ++r) {
      const int row0 = m0 + g * 8 + r;          // D layout: M = g*8 + r
      const int row1 = row0 + 16;
      if (row0 < M) {
        float v = acc0[nt][r] + bv;
        if (relu) v = fmaxf(v, 0.0f);
        if (Yf) Yf[(size_t)row0 * N + col] = v;
        if (Yh) Yh[(size_t)row0 * N + col] = (__bf16)v;
      }
      if (row1 < M) {
        float v = acc1[nt][r] + bv;
        if (relu) v = fmaxf(v, 0.0f);
        if (Yf) Yf[(size_t)row1 * N + col] = v;
        if (Yh) Yh[(size_t)row1 * N + col] = (__bf16)v;
      }
    }
  }
}

// ---------------------------------------------------------------------------
// Fused attention for one (batch, head, 16-row query tile). 1 wave / block.
// Q,K,V bf16 [B*L, D_MODEL] (head h at column h*HDIM); O written bf16.
// fp32 softmax in LDS; probs re-quantized to bf16 for the P@V WMMAs.
// ---------------------------------------------------------------------------
__global__ __launch_bounds__(32)
void efx_attention(const __bf16* __restrict__ Q, const __bf16* __restrict__ Kb,
                   const __bf16* __restrict__ Vb, __bf16* __restrict__ O,
                   int Lq, int Lk, float scale)
{
  __shared__ float  sS[16][SEQ + 4];   // fp32 scores / exp values
  __shared__ __bf16 sP[16][SEQ + 8];   // bf16 probabilities (rows 16B aligned)

  const int lane  = threadIdx.x & 31;
  const int laneN = lane & 15;
  const int g     = lane >> 4;

  const int qtiles = Lq >> 4;
  int t = blockIdx.x;
  const int qt = t % qtiles; t /= qtiles;
  const int h  = t % NHEADS;
  const int b  = t / NHEADS;

  // --- Q fragments (16 x 64 tile = two K-steps of 32) ---
  const __bf16* qp = Q + (size_t)(b * Lq + qt * 16 + laneN) * D_MODEL + h * HDIM;
  v16bf aq[2];
#pragma unroll
  for (int kk = 0; kk < 2; ++kk) {
    const __bf16* qpp = qp + kk * 32 + g * 16;
    aq[kk] = frag_from_halves(*(const v8bf*)qpp, *(const v8bf*)(qpp + 8));
  }

  const v8f vzero = {0.f,0.f,0.f,0.f,0.f,0.f,0.f,0.f};

  // --- scores = Q @ K^T * scale ---
  for (int nt = 0; nt < (Lk >> 4); ++nt) {
    const int krow = b * Lk + nt * 16 + laneN;   // key row for column laneN
    v8f acc = vzero;
#pragma unroll
    for (int kk = 0; kk < 2; ++kk) {
      v16bf bb = *(const v16bf*)(Kb + (size_t)krow * D_MODEL + h * HDIM
                                 + kk * 32 + g * 16);
      acc = wmma_bf16(aq[kk], bb, acc);
    }
#pragma unroll
    for (int r = 0; r < 8; ++r)
      sS[g * 8 + r][nt * 16 + laneN] = acc[r] * scale;
  }
  __syncthreads();

  // --- fp32 softmax, row-cooperative across the wave ---
  for (int r = 0; r < 16; ++r) {
    float m = -3.4e38f;
    for (int c = lane; c < Lk; c += 32) m = fmaxf(m, sS[r][c]);
#pragma unroll
    for (int off = 16; off > 0; off >>= 1) m = fmaxf(m, __shfl_xor(m, off, 32));
    float s = 0.f;
    for (int c = lane; c < Lk; c += 32) {
      float e = __expf(sS[r][c] - m);
      sS[r][c] = e; s += e;
    }
#pragma unroll
    for (int off = 16; off > 0; off >>= 1) s += __shfl_xor(s, off, 32);
    const float inv = 1.0f / s;
    for (int c = lane; c < Lk; c += 32) sP[r][c] = (__bf16)(sS[r][c] * inv);
  }
  __syncthreads();

  // --- O = P @ V  (16 x 64) ---
#pragma unroll
  for (int nt = 0; nt < 4; ++nt) {
    v8f acc = vzero;
    for (int k0 = 0; k0 < Lk; k0 += 32) {
      const __bf16* pp = &sP[laneN][k0 + g * 16];
      v16bf ap = frag_from_halves(*(const v8bf*)pp, *(const v8bf*)(pp + 8));
      v16bf bv;
#pragma unroll
      for (int i = 0; i < 16; ++i)
        bv[i] = Vb[(size_t)(b * Lk + k0 + g * 16 + i) * D_MODEL
                   + h * HDIM + nt * 16 + laneN];
      acc = wmma_bf16(ap, bv, acc);
    }
#pragma unroll
    for (int r = 0; r < 8; ++r)
      O[(size_t)(b * Lq + qt * 16 + g * 8 + r) * D_MODEL
        + h * HDIM + nt * 16 + laneN] = (__bf16)acc[r];
  }
}

// ---------------------------------------------------------------------------
// LayerNorm over D_MODEL (optional residual; optional bf16 mirror output).
// ---------------------------------------------------------------------------
__global__ __launch_bounds__(256)
void efx_ln(const float* __restrict__ Yin, const float* __restrict__ Res,
            const float* __restrict__ Gm, const float* __restrict__ Bt,
            float* __restrict__ Out, __bf16* __restrict__ OutH, float eps)
{
  __shared__ float red[256];
  const int row = blockIdx.x, t = threadIdx.x;
  const size_t base = (size_t)row * D_MODEL;
  float v[3];
#pragma unroll
  for (int j = 0; j < 3; ++j) {
    const int i = t + j * 256;
    v[j] = Yin[base + i] + (Res ? Res[base + i] : 0.f);
  }
  red[t] = v[0] + v[1] + v[2];
  __syncthreads();
  for (int s2 = 128; s2 > 0; s2 >>= 1) {
    if (t < s2) red[t] += red[t + s2];
    __syncthreads();
  }
  const float mean = red[0] * (1.0f / D_MODEL);
  __syncthreads();
  float lv = 0.f;
#pragma unroll
  for (int j = 0; j < 3; ++j) { const float d = v[j] - mean; lv += d * d; }
  red[t] = lv;
  __syncthreads();
  for (int s2 = 128; s2 > 0; s2 >>= 1) {
    if (t < s2) red[t] += red[t + s2];
    __syncthreads();
  }
  const float rstd = rsqrtf(red[0] * (1.0f / D_MODEL) + eps);
#pragma unroll
  for (int j = 0; j < 3; ++j) {
    const int i = t + j * 256;
    const float o = (v[j] - mean) * rstd * Gm[i] + Bt[i];
    if (Out)  Out[base + i]  = o;
    if (OutH) OutH[base + i] = (__bf16)o;
  }
}

// ---------------------------------------------------------------------------
// q = mean over sequence of ce: [B,S,D] -> [B,D]
// ---------------------------------------------------------------------------
__global__ void efx_mean_seq(const float* __restrict__ X, float* __restrict__ Y)
{
  const int idx = blockIdx.x * 256 + threadIdx.x;
  if (idx >= BATCH * D_MODEL) return;
  const int b = idx / D_MODEL, d = idx % D_MODEL;
  float s = 0.f;
  for (int k = 0; k < SEQ; ++k) s += X[((size_t)b * SEQ + k) * D_MODEL + d];
  Y[idx] = s * (1.0f / SEQ);
}

// ---------------------------------------------------------------------------
// Pool attention: 1 query per (b,h). Tiny FLOPs -> VALU kernel, fp32 K/V.
// ---------------------------------------------------------------------------
__global__ __launch_bounds__(256)
void efx_pool_attn(const float* __restrict__ Qp, const float* __restrict__ Kp,
                   const float* __restrict__ Vp, const int* __restrict__ mask,
                   float* __restrict__ Probs, float* __restrict__ Opool)
{
  __shared__ float q[HDIM];
  __shared__ float sc[SEQ];
  __shared__ float red[256];
  const int t = threadIdx.x;
  const int h = blockIdx.x % NHEADS, b = blockIdx.x / NHEADS;
  if (t < HDIM) q[t] = Qp[(size_t)b * D_MODEL + h * HDIM + t];
  __syncthreads();
  for (int kk = t; kk < SEQ; kk += 256) {
    const float* vp = Kp + ((size_t)b * SEQ + kk) * D_MODEL + h * HDIM;
    float dot = 0.f;
#pragma unroll 8
    for (int d = 0; d < HDIM; ++d) dot += q[d] * vp[d];
    dot *= 0.125f;                                  // 1/sqrt(64)
    if (mask[b * SEQ + kk] == 0) dot = -1e9f;       // key padding mask
    sc[kk] = dot;
  }
  __syncthreads();
  float m = -3.4e38f;
  for (int kk = t; kk < SEQ; kk += 256) m = fmaxf(m, sc[kk]);
  red[t] = m;
  __syncthreads();
  for (int s2 = 128; s2 > 0; s2 >>= 1) {
    if (t < s2) red[t] = fmaxf(red[t], red[t + s2]);
    __syncthreads();
  }
  m = red[0];
  __syncthreads();
  float s = 0.f;
  for (int kk = t; kk < SEQ; kk += 256) {
    const float e = __expf(sc[kk] - m);
    sc[kk] = e; s += e;
  }
  red[t] = s;
  __syncthreads();
  for (int s2 = 128; s2 > 0; s2 >>= 1) {
    if (t < s2) red[t] += red[t + s2];
    __syncthreads();
  }
  const float inv = 1.0f / red[0];
  __syncthreads();
  for (int kk = t; kk < SEQ; kk += 256) {
    const float p = sc[kk] * inv;
    sc[kk] = p;
    Probs[((size_t)b * NHEADS + h) * SEQ + kk] = p;
  }
  __syncthreads();
  if (t < HDIM) {
    float acc = 0.f;
    for (int kk = 0; kk < SEQ; ++kk)
      acc += sc[kk] * Vp[((size_t)b * SEQ + kk) * D_MODEL + h * HDIM + t];
    Opool[(size_t)b * D_MODEL + h * HDIM + t] = acc;
  }
}

// attention_weights = mean over heads of pool probabilities -> [B, S]
__global__ void efx_head_mean(const float* __restrict__ Probs,
                              float* __restrict__ AW)
{
  const int idx = blockIdx.x * 256 + threadIdx.x;
  if (idx >= BATCH * SEQ) return;
  const int b = idx / SEQ, s = idx % SEQ;
  float acc = 0.f;
  for (int h = 0; h < NHEADS; ++h)
    acc += Probs[((size_t)b * NHEADS + h) * SEQ + s];
  AW[idx] = acc * (1.0f / NHEADS);
}

// ---------------------------------------------------------------------------
extern "C" void kernel_launch(void* const* d_in, const int* in_sizes, int n_in,
                              void* d_out, int out_size, void* d_ws, size_t ws_size,
                              hipStream_t stream)
{
  (void)in_sizes; (void)n_in; (void)out_size; (void)ws_size;

  const float* hs    = (const float*)d_in[0];
  const int*   am    = (const int*)  d_in[1];
  const float* ctx   = (const float*)d_in[2];
  const float* sa_w[4] = { (const float*)d_in[3], (const float*)d_in[5],
                           (const float*)d_in[7], (const float*)d_in[9] };
  const float* sa_bq = (const float*)d_in[4],  *sa_bk = (const float*)d_in[6];
  const float* sa_bv = (const float*)d_in[8],  *sa_bo = (const float*)d_in[10];
  const float* sa_g  = (const float*)d_in[11], *sa_b  = (const float*)d_in[12];
  const float* ca_w[4] = { (const float*)d_in[13], (const float*)d_in[15],
                           (const float*)d_in[17], (const float*)d_in[19] };
  const float* ca_bq = (const float*)d_in[14], *ca_bk = (const float*)d_in[16];
  const float* ca_bv = (const float*)d_in[18], *ca_bo = (const float*)d_in[20];
  const float* ca_g  = (const float*)d_in[21], *ca_b  = (const float*)d_in[22];
  const float* pool_w_in = (const float*)d_in[23], *pool_b_in = (const float*)d_in[24];
  const float* pool_wo   = (const float*)d_in[25], *pool_bo   = (const float*)d_in[26];
  const float* fus_w1 = (const float*)d_in[27], *fus_b1 = (const float*)d_in[28];
  const float* fus_w2 = (const float*)d_in[29], *fus_b2 = (const float*)d_in[30];
  const float* fus_g  = (const float*)d_in[31], *fus_b  = (const float*)d_in[32];

  // Output layout: features | ce | attention_weights | pooled
  float* out        = (float*)d_out;
  float* o_features = out;
  float* o_ce       = out + (size_t)BATCH * D_MODEL;
  float* o_aw       = o_ce + (size_t)BATCH * SEQ * D_MODEL;
  float* o_pooled   = o_aw + (size_t)BATCH * SEQ;

  // ---- workspace carving (64B-aligned allocations) ----
  const size_t SLOT = (size_t)BATCH * SEQ * D_MODEL;    // 12,582,912
  const size_t WMAT = (size_t)D_MODEL * D_MODEL;        // 589,824
  char* wsb = (char*)d_ws;
  auto alloc_f32 = [&](size_t n) {
    float* p = (float*)wsb; wsb += ((n * 4 + 63) & ~(size_t)63); return p;
  };
  auto alloc_bf = [&](size_t n) {
    __bf16* p = (__bf16*)wsb; wsb += ((n * 2 + 63) & ~(size_t)63); return p;
  };

  float*  f0      = alloc_f32(SLOT);            // proj tmp / pool K
  float*  f1      = alloc_f32(SLOT);            // enh f32 / pool V
  __bf16* hb0     = alloc_bf(SLOT);             // hs_bf -> enh_bf -> ce_bf
  __bf16* hb1     = alloc_bf(SLOT);             // Q
  __bf16* hb2     = alloc_bf(SLOT);             // K
  __bf16* hb3     = alloc_bf(SLOT);             // V
  __bf16* hb4     = alloc_bf(SLOT);             // attention O
  __bf16* ctx_bf  = alloc_bf((size_t)BATCH * CTX * D_MODEL);
  __bf16* sa_wb[4], *ca_wb[4];
  for (int i = 0; i < 4; ++i) sa_wb[i] = alloc_bf(WMAT);
  for (int i = 0; i < 4; ++i) ca_wb[i] = alloc_bf(WMAT);
  __bf16* pool_win_bf = alloc_bf(3 * WMAT);
  __bf16* pool_wo_bf  = alloc_bf(WMAT);
  __bf16* fus_w1_bf   = alloc_bf(2 * WMAT);
  __bf16* fus_w2_bf   = alloc_bf(2 * WMAT);
  float*  qmean    = alloc_f32((size_t)BATCH * D_MODEL);
  float*  poolQ    = alloc_f32((size_t)BATCH * D_MODEL);
  float*  probs    = alloc_f32((size_t)BATCH * NHEADS * SEQ);
  float*  poolO    = alloc_f32((size_t)BATCH * D_MODEL);
  float*  yfus     = alloc_f32((size_t)BATCH * D_MODEL);
  __bf16* qmean_bf = alloc_bf((size_t)BATCH * D_MODEL);
  __bf16* poolO_bf = alloc_bf((size_t)BATCH * D_MODEL);
  __bf16* pooled_bf= alloc_bf((size_t)BATCH * D_MODEL);
  __bf16* h1h      = alloc_bf((size_t)BATCH * 2 * D_MODEL);

  auto cvt = [&](const float* X, __bf16* Y, size_t n) {
    efx_cvt_bf16<<<dim3((unsigned)((n + 255) / 256)), dim3(256), 0, stream>>>(
        X, Y, (int)n);
  };
  auto gemm = [&](const __bf16* X, const __bf16* W, const float* Bv,
                  float* Yf, __bf16* Yh, int M, int N, int K, int relu) {
    dim3 grid(N / 64, (M + 127) / 128);
    efx_gemm_bias<<<grid, dim3(128), 0, stream>>>(X, W, Bv, Yf, Yh, M, N, K, relu);
  };

  const int MS = BATCH * SEQ;   // 16384
  const int MC = BATCH * CTX;   // 4096
  const float scale = 0.125f;   // 1/sqrt(HDIM)

  // ---- one-time bf16 staging of activations + all weights ----
  cvt(hs,  hb0,    SLOT);
  cvt(ctx, ctx_bf, (size_t)BATCH * CTX * D_MODEL);
  for (int i = 0; i < 4; ++i) cvt(sa_w[i], sa_wb[i], WMAT);
  for (int i = 0; i < 4; ++i) cvt(ca_w[i], ca_wb[i], WMAT);
  cvt(pool_w_in, pool_win_bf, 3 * WMAT);
  cvt(pool_wo,   pool_wo_bf,  WMAT);
  cvt(fus_w1,    fus_w1_bf,   2 * WMAT);
  cvt(fus_w2,    fus_w2_bf,   2 * WMAT);

  // ---- self-attention block ----
  gemm(hb0, sa_wb[0], sa_bq, nullptr, hb1, MS, D_MODEL, D_MODEL, 0);
  gemm(hb0, sa_wb[1], sa_bk, nullptr, hb2, MS, D_MODEL, D_MODEL, 0);
  gemm(hb0, sa_wb[2], sa_bv, nullptr, hb3, MS, D_MODEL, D_MODEL, 0);
  efx_attention<<<dim3(BATCH * NHEADS * (SEQ / 16)), dim3(32), 0, stream>>>(
      hb1, hb2, hb3, hb4, SEQ, SEQ, scale);
  gemm(hb4, sa_wb[3], sa_bo, f0, nullptr, MS, D_MODEL, D_MODEL, 0);
  efx_ln<<<dim3(MS), dim3(256), 0, stream>>>(f0, hs, sa_g, sa_b, f1, hb0, 1e-5f);

  // ---- cross-attention block ----
  gemm(hb0, ca_wb[0], ca_bq, nullptr, hb1, MS, D_MODEL, D_MODEL, 0);
  gemm(ctx_bf, ca_wb[1], ca_bk, nullptr, hb2, MC, D_MODEL, D_MODEL, 0);
  gemm(ctx_bf, ca_wb[2], ca_bv, nullptr, hb3, MC, D_MODEL, D_MODEL, 0);
  efx_attention<<<dim3(BATCH * NHEADS * (SEQ / 16)), dim3(32), 0, stream>>>(
      hb1, hb2, hb3, hb4, SEQ, CTX, scale);
  gemm(hb4, ca_wb[3], ca_bo, f0, nullptr, MS, D_MODEL, D_MODEL, 0);
  efx_ln<<<dim3(MS), dim3(256), 0, stream>>>(f0, f1, ca_g, ca_b, o_ce, hb0, 1e-5f);

  // ---- position-aware attention pooling ----
  efx_mean_seq<<<dim3((BATCH * D_MODEL + 255) / 256), dim3(256), 0, stream>>>(
      o_ce, qmean);
  cvt(qmean, qmean_bf, (size_t)BATCH * D_MODEL);
  gemm(qmean_bf, pool_win_bf, pool_b_in, poolQ, nullptr,
       BATCH, D_MODEL, D_MODEL, 0);
  gemm(hb0, pool_win_bf + WMAT, pool_b_in + D_MODEL, f0, nullptr,
       MS, D_MODEL, D_MODEL, 0);                               // pool K
  gemm(hb0, pool_win_bf + 2 * WMAT, pool_b_in + 2 * D_MODEL, f1, nullptr,
       MS, D_MODEL, D_MODEL, 0);                               // pool V
  efx_pool_attn<<<dim3(BATCH * NHEADS), dim3(256), 0, stream>>>(
      poolQ, f0, f1, am, probs, poolO);
  efx_head_mean<<<dim3((BATCH * SEQ + 255) / 256), dim3(256), 0, stream>>>(
      probs, o_aw);
  cvt(poolO, poolO_bf, (size_t)BATCH * D_MODEL);
  gemm(poolO_bf, pool_wo_bf, pool_bo, o_pooled, pooled_bf,
       BATCH, D_MODEL, D_MODEL, 0);

  // ---- feature fusion MLP + LayerNorm(1e-12) ----
  gemm(pooled_bf, fus_w1_bf, fus_b1, nullptr, h1h, BATCH, 2 * D_MODEL, D_MODEL, 1);
  gemm(h1h, fus_w2_bf, fus_b2, yfus, nullptr, BATCH, D_MODEL, 2 * D_MODEL, 0);
  efx_ln<<<dim3(BATCH), dim3(256), 0, stream>>>(yfus, nullptr, fus_g, fus_b,
                                                o_features, nullptr, 1e-12f);
}
